// RWKVTimeMix_24756191494367
// MI455X (gfx1250) — matrix operand order repaired and checked by reading, not compiled, not run on hardware
//
#include <hip/hip_runtime.h>

// ---------------------------------------------------------------------------
// RWKV TimeMix for MI455X (gfx1250, wave32, WMMA, async global->LDS staging).
//   k = x @ Wk^T ; r = x @ Wr^T ; v = r          (bf16 WMMA GEMMs, f32 accum)
//   y = wkv_scan(w=-exp(td), u=tf, k, v)         (serial-in-T scan kernel)
//   out = sigmoid(r) * y                         (fused into scan, bf16 out)
//   result = out @ Wo^T                          (bf16 WMMA GEMM, f32 out)
// All GEMMs: M x 1024 x 1024 (N = K = 1024 compile-time constants).
// ---------------------------------------------------------------------------

typedef __attribute__((ext_vector_type(16))) __bf16 v16bf;
typedef __attribute__((ext_vector_type(8)))  __bf16 v8bf;
typedef __attribute__((ext_vector_type(8)))  float  v8f;
typedef __attribute__((ext_vector_type(4)))  int    v4i;

#define BM 256          // workgroup M tile
#define BN 128          // workgroup N tile
#define BK 32           // K step (one bf16 WMMA K)
#define GN 1024         // GEMM N (compile-time)
#define GK 1024         // GEMM K (compile-time)
#define LDST 40         // padded LDS row stride (bf16 elems, 80B) - conflict-free

union Frag { v16bf v; v8bf h[2]; };

__device__ __forceinline__ unsigned short f2bf(float f) {
  unsigned int u = __float_as_uint(f);
  return (unsigned short)((u + 0x7FFFu + ((u >> 16) & 1u)) >> 16);  // RNE
}

// ---- gfx1250 async global->LDS staging (ASYNCcnt path), sync fallback ------
#if __has_builtin(__builtin_amdgcn_global_load_async_to_lds_b128)
__device__ __forceinline__ void cp16_async(const unsigned short* g,
                                           unsigned short* s) {
  typedef __attribute__((address_space(1))) v4i* gv4i_p;
  typedef __attribute__((address_space(3))) v4i* sv4i_p;
  void* gnc = (void*)g;  // drop const, then addrspace+type cast
  __builtin_amdgcn_global_load_async_to_lds_b128(
      (gv4i_p)gnc, (sv4i_p)(void*)s, 0, 0);
}
#else
__device__ __forceinline__ void cp16_async(const unsigned short* g,
                                           unsigned short* s) {
  *(uint4*)s = *(const uint4*)g;          // fallback: load + ds_store
}
#endif

__device__ __forceinline__ void wait_async0() {
#if __has_builtin(__builtin_amdgcn_s_wait_asynccnt)
  __builtin_amdgcn_s_wait_asynccnt(0);
#else
  asm volatile("s_wait_asynccnt 0x0" ::: "memory");
#endif
}

// -------------------------------- fp32 -> bf16 ------------------------------
__global__ __launch_bounds__(256) void cvt_f32_bf16_kernel(
    const float* __restrict__ in, unsigned short* __restrict__ out, int n4) {
  int i = blockIdx.x * blockDim.x + threadIdx.x;
  if (i < n4) {
    const float4 f = ((const float4*)in)[i];
    uint2 o;
    o.x = (unsigned int)f2bf(f.x) | ((unsigned int)f2bf(f.y) << 16);
    o.y = (unsigned int)f2bf(f.z) | ((unsigned int)f2bf(f.w) << 16);
    ((uint2*)out)[i] = o;
  }
}

// ------------------------- bf16 WMMA GEMM: C = A * W^T ----------------------
// A: [M,1024] bf16 row-major; W: [1024,1024] bf16 row-major (K contiguous)
// C: [M,1024] f32. 8 waves; each wave computes a 64x64 patch (4x4 WMMA tiles).
__global__ __launch_bounds__(256, 1) void gemm_bf16_wmma_kernel(
    const unsigned short* __restrict__ A,
    const unsigned short* __restrict__ W,
    float* __restrict__ C, int M) {
  __shared__ unsigned short ldsA[2][BM * LDST];   // 2 x 20 KB
  __shared__ unsigned short ldsB[2][BN * LDST];   // 2 x 10 KB

  const int tid   = threadIdx.x;
  const int lane  = tid & 31;
  const int wave  = tid >> 5;          // 8 waves (wave32)
  const int waveM = wave >> 1;         // 0..3 -> 64-row slab
  const int waveN = wave & 1;          // 0..1 -> 64-col slab
  const int m16   = lane & 15;
  const int hi    = lane >> 4;

  const int blockM = blockIdx.y * BM;
  const int blockN = blockIdx.x * BN;

  // staging: thread moves 4 A-chunks + 2 B-chunks (16B each) per K tile
  const int r4 = tid >> 2;             // 0..63
  const int cg = (tid & 3) * 8;        // bf16 elem offset in 32-wide K tile

  const unsigned short* gA[4];
  const unsigned short* gB[2];
#pragma unroll
  for (int i = 0; i < 4; ++i)
    gA[i] = A + (size_t)(blockM + r4 + 64 * i) * GK + cg;
#pragma unroll
  for (int i = 0; i < 2; ++i)
    gB[i] = W + (size_t)(blockN + r4 + 64 * i) * GK + cg;

  const int sAo = r4 * LDST + cg;      // LDS elem offsets (per 64-row slab)
  const int sBo = r4 * LDST + cg;

  v8f acc[4][4] = {};

  // prologue: stage K-tile 0 into buffer 0
#pragma unroll
  for (int i = 0; i < 4; ++i)
    cp16_async(gA[i], &ldsA[0][sAo + i * 64 * LDST]);
#pragma unroll
  for (int i = 0; i < 2; ++i)
    cp16_async(gB[i], &ldsB[0][sBo + i * 64 * LDST]);
  wait_async0();
  __syncthreads();

  const int KT = GK / BK;              // 32
  for (int kt = 0; kt < KT; ++kt) {
    const int cur = kt & 1;
    const bool more = (kt + 1) < KT;

    if (more) {                        // stage next tile into other buffer
      const int nxt = cur ^ 1;
      const size_t koff = (size_t)(kt + 1) * BK;
#pragma unroll
      for (int i = 0; i < 4; ++i)
        cp16_async(gA[i] + koff, &ldsA[nxt][sAo + i * 64 * LDST]);
#pragma unroll
      for (int i = 0; i < 2; ++i)
        cp16_async(gB[i] + koff, &ldsB[nxt][sBo + i * 64 * LDST]);
    }
    if (kt + 2 < KT) {                 // gfx1250 global_prefetch_b8 path
      __builtin_prefetch(gA[0] + (size_t)(kt + 2) * BK, 0, 3);
      __builtin_prefetch(gB[0] + (size_t)(kt + 2) * BK, 0, 3);
    }

    // A frags: lane = row (m16); halves 0-7 = K[hi*8..+7], 8-15 = K[16+hi*8..]
    Frag afr[4];
#pragma unroll
    for (int mi = 0; mi < 4; ++mi) {
      const unsigned short* p =
          &ldsA[cur][(waveM * 64 + mi * 16 + m16) * LDST + hi * 8];
      afr[mi].h[0] = *(const v8bf*)(p);
      afr[mi].h[1] = *(const v8bf*)(p + 16);
    }
    // B frags: lane = col (m16); halves 0-7 = K[hi*8..+7], 8-15 = K[16+hi*8..]
    Frag bfr[4];
#pragma unroll
    for (int ni = 0; ni < 4; ++ni) {
      const unsigned short* p =
          &ldsB[cur][(waveN * 64 + ni * 16 + m16) * LDST + hi * 8];
      bfr[ni].h[0] = *(const v8bf*)(p);
      bfr[ni].h[1] = *(const v8bf*)(p + 16);
    }

#pragma unroll
    for (int mi = 0; mi < 4; ++mi)
#pragma unroll
      for (int ni = 0; ni < 4; ++ni)
        acc[mi][ni] = __builtin_amdgcn_wmma_f32_16x16x32_bf16(
            false, afr[mi].v, false, bfr[ni].v, (short)0, acc[mi][ni],
            false, false);

    if (more) wait_async0();
    __syncthreads();
  }

  // C/D layout: VGPR j -> row (hi*8 + j), col = m16 ; 32-bit offsets (N=1024)
  const unsigned int obase =
      (unsigned int)(blockM + waveM * 64 + hi * 8) * GN +
      (unsigned int)(blockN + waveN * 64 + m16);
#pragma unroll
  for (int mi = 0; mi < 4; ++mi)
#pragma unroll
    for (int ni = 0; ni < 4; ++ni) {
      unsigned int o = obase + (unsigned int)(mi * 16) * GN + ni * 16;
#pragma unroll
      for (int j = 0; j < 8; ++j)
        C[o + (unsigned int)j * GN] = acc[mi][ni][j];
    }
}

// ----------------------------- WKV scan (serial T) --------------------------
__global__ __launch_bounds__(256) void wkv_scan_kernel(
    const float* __restrict__ Kin, const float* __restrict__ Rin,
    const float* __restrict__ time_decay, const float* __restrict__ time_first,
    unsigned short* __restrict__ out_bf, int Bq, int T, int Cc) {
  const int idx = blockIdx.x * blockDim.x + threadIdx.x;
  if (idx >= Bq * Cc) return;
  const int b = idx / Cc;
  const int c = idx - b * Cc;

  const float w = -__expf(time_decay[c]);
  const float u = time_first[c];

  float a = 0.f, bden = 0.f, pp = -1e38f;
  size_t base = (size_t)b * T * Cc + c;
  for (int t = 0; t < T; ++t, base += Cc) {
    const float kt = Kin[base];
    const float vt = Rin[base];        // v = r (faithful to source)
    const float ww = u + kt;
    const float q  = fmaxf(pp, ww);
    const float e1 = __expf(pp - q);
    const float e2 = __expf(ww - q);
    const float y  = (e1 * a + e2 * vt) / (e1 * bden + e2);
    const float ww2 = pp + w;
    const float q2  = fmaxf(ww2, kt);
    const float e1b = __expf(ww2 - q2);
    const float e2b = __expf(kt - q2);
    a    = e1b * a + e2b * vt;
    bden = e1b * bden + e2b;
    pp   = q2;
    const float sg = 1.0f / (1.0f + __expf(-vt));   // sigmoid(r)
    out_bf[base] = f2bf(sg * y);
  }
}

// ---------------------------------------------------------------------------
extern "C" void kernel_launch(void* const* d_in, const int* in_sizes, int n_in,
                              void* d_out, int out_size, void* d_ws, size_t ws_size,
                              hipStream_t stream) {
  const float* x  = (const float*)d_in[0];
  const float* Wk = (const float*)d_in[1];
  const float* Wr = (const float*)d_in[2];
  const float* Wo = (const float*)d_in[3];
  const float* td = (const float*)d_in[4];
  const float* tf = (const float*)d_in[5];

  const int Cc = in_sizes[4];          // 1024
  const int BT = in_sizes[0] / Cc;     // 16384
  const int T  = 1024;
  const int Bq = BT / T;               // 16

  char* ws = (char*)d_ws;
  unsigned short* x_bf  = (unsigned short*)ws; ws += (size_t)BT * Cc * 2;
  unsigned short* wk_bf = (unsigned short*)ws; ws += (size_t)Cc * Cc * 2;
  unsigned short* wr_bf = (unsigned short*)ws; ws += (size_t)Cc * Cc * 2;
  unsigned short* wo_bf = (unsigned short*)ws; ws += (size_t)Cc * Cc * 2;
  float* k_f = (float*)ws;             ws += (size_t)BT * Cc * 4;
  float* r_f = (float*)ws;             ws += (size_t)BT * Cc * 4;
  unsigned short* out_bf = x_bf;       // x_bf is dead after the two input GEMMs

  const int n4 = (BT * Cc) / 4;
  const int w4 = (Cc * Cc) / 4;
  cvt_f32_bf16_kernel<<<(n4 + 255) / 256, 256, 0, stream>>>(x,  x_bf,  n4);
  cvt_f32_bf16_kernel<<<(w4 + 255) / 256, 256, 0, stream>>>(Wk, wk_bf, w4);
  cvt_f32_bf16_kernel<<<(w4 + 255) / 256, 256, 0, stream>>>(Wr, wr_bf, w4);
  cvt_f32_bf16_kernel<<<(w4 + 255) / 256, 256, 0, stream>>>(Wo, wo_bf, w4);

  dim3 grid(Cc / BN, BT / BM);         // (8, 64)
  gemm_bf16_wmma_kernel<<<grid, 256, 0, stream>>>(x_bf, wk_bf, k_f, BT);
  gemm_bf16_wmma_kernel<<<grid, 256, 0, stream>>>(x_bf, wr_bf, r_f, BT);

  wkv_scan_kernel<<<(Bq * Cc + 255) / 256, 256, 0, stream>>>(
      k_f, r_f, td, tf, out_bf, Bq, T, Cc);

  gemm_bf16_wmma_kernel<<<grid, 256, 0, stream>>>(
      out_bf, wo_bf, (float*)d_out, BT);
}
